// BiasedMHA_1159641170466
// MI455X (gfx1250) — matrix operand order repaired and checked by reading, not compiled, not run on hardware
//
#include <hip/hip_runtime.h>

typedef __bf16 bf16;
typedef bf16 v16bf __attribute__((ext_vector_type(16)));
typedef bf16 v8bf  __attribute__((ext_vector_type(8)));
typedef float v8f  __attribute__((ext_vector_type(8)));
typedef unsigned int v4u __attribute__((ext_vector_type(4)));
typedef int v4i __attribute__((ext_vector_type(4)));
typedef int v8i __attribute__((ext_vector_type(8)));

#define B_    4
#define N_    1024
#define FEAT_ 512
#define H_    8
#define HD_   64

static __device__ __forceinline__ bf16 f2bf(float f) {
  unsigned u = __float_as_uint(f);
  unsigned r = (u + 0x7fffu + ((u >> 16) & 1u)) >> 16;
  unsigned short s = (unsigned short)r;
  return __builtin_bit_cast(bf16, s);
}

static __device__ __forceinline__ v16bf cat8(v8bf lo, v8bf hi) {
  return __builtin_shufflevector(lo, hi, 0,1,2,3,4,5,6,7,8,9,10,11,12,13,14,15);
}

static __device__ __forceinline__ v8f wmma_bf16(v16bf a, v16bf b, v8f c) {
  // D = A(16x32 bf16) * B(32x16 bf16) + C(16x16 f32)
  return __builtin_amdgcn_wmma_f32_16x16x32_bf16(false, a, false, b, (short)0, c, false, false);
}

// ---- Tensor Data Mover: 2-D tile of 2-byte elements, global -> LDS ----------
// D# per CDNA5 ISA:
//  g0: [1:0]=count=1, [63:32]=lds_addr(bytes), [120:64]=global_addr, [127:126]=type=2
//  g1: [17:16]=data_size=1 (2B), dim0/dim1 tensor+tile sizes, dim0 stride
static __device__ __forceinline__ void tdm_load_2d_bf16(unsigned lds_off, const void* gptr,
                                                        unsigned tile_d0, unsigned tile_d1,
                                                        unsigned tensor_d0, unsigned tensor_d1,
                                                        unsigned stride0_elems) {
  unsigned long long ga = (unsigned long long)(uintptr_t)gptr;
  v4u g0;
  g0[0] = 1u;                                               // count=1, user D#
  g0[1] = lds_off;                                          // lds_addr (bytes)
  g0[2] = (unsigned)(ga & 0xffffffffu);                     // global_addr lo
  g0[3] = (unsigned)((ga >> 32) & 0x01ffffffu) | (2u << 30); // global_addr hi | type=2
  v8i g1;
  g1[0] = 0x00010000;                                       // data_size=1 (2 bytes)
  g1[1] = (int)((tensor_d0 & 0xffffu) << 16);               // tensor_dim0[15:0]
  g1[2] = (int)(((tensor_d0 >> 16) & 0xffffu) | ((tensor_d1 & 0xffffu) << 16));
  g1[3] = (int)(((tensor_d1 >> 16) & 0xffffu) | ((tile_d0 & 0xffffu) << 16));
  g1[4] = (int)(tile_d1 & 0xffffu);                         // tile_dim1, tile_dim2=0
  g1[5] = (int)stride0_elems;                               // tensor_dim0_stride[31:0]
  g1[6] = 0;
  g1[7] = 0;
  v4i z4 = {0, 0, 0, 0};
#if __clang_major__ >= 23
  v8i z8 = {0, 0, 0, 0, 0, 0, 0, 0};
  __builtin_amdgcn_tensor_load_to_lds(g0, g1, z4, z4, z8, 0);
#else
  __builtin_amdgcn_tensor_load_to_lds(g0, g1, z4, z4, 0);
#endif
}

// ---------------------------------------------------------------- convert f32 -> bf16
__global__ void cvt_kernel(const float* __restrict__ src, bf16* __restrict__ dst, int n) {
  int i = blockIdx.x * blockDim.x + threadIdx.x;
  if (i < n) dst[i] = f2bf(src[i]);
}

// ---------------------------------------------------------------- QKV projection
// 4 waves / workgroup; weight tile (16 x 512 bf16) staged once via TDM into LDS,
// each wave owns one 16-row M tile. NT gemm: C[row,f] = sum_c nd[row,c]*W[f,c] + bias[f].
__global__ void qkv_proj_kernel(const bf16* __restrict__ nd,
                                const bf16* __restrict__ wq, const bf16* __restrict__ wk,
                                const bf16* __restrict__ wv,
                                const float* __restrict__ bq, const float* __restrict__ bk,
                                const float* __restrict__ bv,
                                bf16* __restrict__ Q, bf16* __restrict__ K, bf16* __restrict__ Vt) {
  __shared__ bf16 Bsh[16 * FEAT_];   // 16 KB weight tile: rows f = nt*16..+15

  const int tid = threadIdx.x;
  const int lane = tid & 31, wave = tid >> 5;
  const int half = lane >> 4, lr = lane & 15;
  const int mt = blockIdx.x * 4 + wave, nt = blockIdx.y, mode = blockIdx.z;
  const bf16* w    = (mode == 0) ? wq : (mode == 1) ? wk : wv;
  const float* bia = (mode == 0) ? bq : (mode == 1) ? bk : bv;

  if (wave == 0) {   // wave-uniform branch: EXEC all-ones inside
    tdm_load_2d_bf16((unsigned)(uintptr_t)Bsh, w + (size_t)nt * 16 * FEAT_,
                     FEAT_, 16, FEAT_, 16, FEAT_);
    __builtin_amdgcn_s_wait_tensorcnt(0);
  }
  __syncthreads();

  const bf16* arow = nd + (size_t)(mt * 16 + lr) * FEAT_;   // A: global, row = M
  const bf16* brow = Bsh + lr * FEAT_;                      // B: LDS, row = N
  v8f c = {};
#pragma unroll 4
  for (int kk = 0; kk < FEAT_ / 32; ++kk) {
    const int kb = kk * 32;
    v16bf a = cat8(*(const v8bf*)(arow + kb + 8 * half),
                   *(const v8bf*)(arow + kb + 16 + 8 * half));
    v16bf b = cat8(*(const v8bf*)(brow + kb + 8 * half),
                   *(const v8bf*)(brow + kb + 16 + 8 * half));
    c = wmma_bf16(a, b, c);
  }
  const int f = nt * 16 + lr, h = f >> 6, d = f & (HD_ - 1);
  const float bval = bia[f];
#pragma unroll
  for (int v = 0; v < 8; ++v) {
    const int m    = v + 8 * half;
    const int grow = mt * 16 + m;          // global row in [0, B*N)
    const int bb   = grow >> 10, i = grow & (N_ - 1);
    const float val = c[v] + bval;
    if (mode == 0)
      Q[(((size_t)bb * H_ + h) * N_ + i) * HD_ + d] = f2bf(val * 0.125f); // HD^-0.5
    else if (mode == 1)
      K[(((size_t)bb * H_ + h) * N_ + i) * HD_ + d] = f2bf(val);
    else
      Vt[(((size_t)bb * H_ + h) * HD_ + d) * N_ + i] = f2bf(val);        // transposed
  }
}

// ---------------------------------------------------------------- fused attention
// One workgroup (256 threads = 8 waves) per (b, h, 16-query tile).
__global__ void attn_kernel(const bf16* __restrict__ Q, const bf16* __restrict__ K,
                            const bf16* __restrict__ Vt,
                            const float* __restrict__ bias, const unsigned char* __restrict__ mask,
                            bf16* __restrict__ O) {
  __shared__ float S[16 * N_];   // 64 KB: 16 query rows x 1024 key cols (scores -> probs)

  const int tid = threadIdx.x;
  const int lane = tid & 31, wave = tid >> 5;
  const int half = lane >> 4, lr = lane & 15;
  const int bid = blockIdx.x;
  const int itile = bid & 63, bh = bid >> 6;
  const int h = bh & (H_ - 1), b = bh >> 3;
  const int i0 = itile * 16;

  // Q tile A-fragments (HD=64 -> two K=32 fragments); same for every wave.
  const bf16* qrow = Q + ((size_t)bh * N_ + i0 + lr) * HD_;
  const v16bf aq0 = cat8(*(const v8bf*)(qrow + 8 * half),
                         *(const v8bf*)(qrow + 16 + 8 * half));
  const v16bf aq1 = cat8(*(const v8bf*)(qrow + 32 + 8 * half),
                         *(const v8bf*)(qrow + 48 + 8 * half));

  // Phase 1: S = Q*K^T + bias, masked -> LDS. Waves stride over the 64 key tiles.
  for (int jt = wave; jt < N_ / 16; jt += 8) {
    const int j0 = jt * 16;
    // prefetch next tile's bias/mask stream (strided 32B granules -> hide latency)
    const int jn = jt + 8;
    if (jn < N_ / 16) {
      const size_t prow = (size_t)(b * N_ + i0 + lr) * N_;
      __builtin_prefetch(bias + (prow + jn * 16 + 8 * half) * H_ + h, 0, 1);
      __builtin_prefetch(bias + (prow + jn * 16 + 4 + 8 * half) * H_ + h, 0, 1);
      __builtin_prefetch(mask + (prow + jn * 16 + 8 * half) * H_ + h, 0, 1);
    }
    const bf16* krow = K + ((size_t)bh * N_ + j0 + lr) * HD_;  // B operand: col j, K=d
    v16bf bk0 = cat8(*(const v8bf*)(krow + 8 * half),
                     *(const v8bf*)(krow + 16 + 8 * half));
    v16bf bk1 = cat8(*(const v8bf*)(krow + 32 + 8 * half),
                     *(const v8bf*)(krow + 48 + 8 * half));
    v8f c = {};
    c = wmma_bf16(aq0, bk0, c);
    c = wmma_bf16(aq1, bk1, c);
    const int j = j0 + lr;
#pragma unroll
    for (int v = 0; v < 8; ++v) {
      const int m = v + 8 * half;
      const size_t idx = (((size_t)(b * N_ + i0 + m)) * N_ + j) * H_ + h;
      float s = c[v] + bias[idx];
      if (mask[idx]) s = -__builtin_inff();
      S[m * N_ + j] = s;
    }
  }
  __syncthreads();

  // Phase 2: row softmax. Each half-wave (16 lanes) owns row r = 2*wave + half.
  {
    const int r = 2 * wave + half;
    float* Srow = S + r * N_;
    float mx = -__builtin_inff();
    for (int c0 = lr; c0 < N_; c0 += 16) mx = fmaxf(mx, Srow[c0]);
#pragma unroll
    for (int o = 8; o >= 1; o >>= 1) mx = fmaxf(mx, __shfl_xor(mx, o, 32));
    float sum = 0.f;
    for (int c0 = lr; c0 < N_; c0 += 16) {
      float e = __expf(Srow[c0] - mx);
      Srow[c0] = e;
      sum += e;
    }
#pragma unroll
    for (int o = 8; o >= 1; o >>= 1) sum += __shfl_xor(sum, o, 32);
    const float rinv = 1.0f / sum;
    for (int c0 = lr; c0 < N_; c0 += 16) Srow[c0] *= rinv;   // normalize in place
  }
  __syncthreads();

  // Phase 3: out = P * V. Waves 0..3 each own a 16-wide d-tile, chain K over all keys.
  if (wave < 4) {
    const int d0 = wave * 16;
    const bf16* vrow = Vt + ((size_t)bh * HD_ + d0 + lr) * N_;  // B operand: col d, K=j
    v8f c = {};
    for (int jc = 0; jc < N_ / 32; ++jc) {
      const int jb = jc * 32;
      const float* srow = S + lr * N_ + jb;      // A operand: row i, K=j (f32 -> bf16)
      v8f lo = *(const v8f*)(srow + 8 * half);
      v8f hi = *(const v8f*)(srow + 16 + 8 * half);
      v16bf a;
#pragma unroll
      for (int e = 0; e < 8; ++e) { a[e] = f2bf(lo[e]); a[e + 8] = f2bf(hi[e]); }
      v16bf bb = cat8(*(const v8bf*)(vrow + jb + 8 * half),
                      *(const v8bf*)(vrow + jb + 16 + 8 * half));
      c = wmma_bf16(a, bb, c);
    }
#pragma unroll
    for (int v = 0; v < 8; ++v) {
      const int m = v + 8 * half;
      O[((size_t)b * N_ + i0 + m) * FEAT_ + h * HD_ + d0 + lr] = f2bf(c[v]);
    }
  }
}

// ---------------------------------------------------------------- output projection (f32 out)
// Same structure as qkv_proj: 4 waves share one TDM-staged Wo tile.
__global__ void out_proj_kernel(const bf16* __restrict__ Oin, const bf16* __restrict__ wo,
                                const float* __restrict__ bo, float* __restrict__ out) {
  __shared__ bf16 Bsh[16 * FEAT_];

  const int tid = threadIdx.x;
  const int lane = tid & 31, wave = tid >> 5;
  const int half = lane >> 4, lr = lane & 15;
  const int mt = blockIdx.x * 4 + wave, nt = blockIdx.y;

  if (wave == 0) {
    tdm_load_2d_bf16((unsigned)(uintptr_t)Bsh, wo + (size_t)nt * 16 * FEAT_,
                     FEAT_, 16, FEAT_, 16, FEAT_);
    __builtin_amdgcn_s_wait_tensorcnt(0);
  }
  __syncthreads();

  const bf16* arow = Oin + (size_t)(mt * 16 + lr) * FEAT_;
  const bf16* brow = Bsh + lr * FEAT_;
  v8f c = {};
#pragma unroll 4
  for (int kk = 0; kk < FEAT_ / 32; ++kk) {
    const int kb = kk * 32;
    v16bf a = cat8(*(const v8bf*)(arow + kb + 8 * half),
                   *(const v8bf*)(arow + kb + 16 + 8 * half));
    v16bf b = cat8(*(const v8bf*)(brow + kb + 8 * half),
                   *(const v8bf*)(brow + kb + 16 + 8 * half));
    c = wmma_bf16(a, b, c);
  }
  const int f = nt * 16 + lr;
  const float bval = bo[f];
#pragma unroll
  for (int v = 0; v < 8; ++v) {
    const int m = v + 8 * half;
    out[(size_t)(mt * 16 + m) * FEAT_ + f] = c[v] + bval;
  }
}

// ---------------------------------------------------------------- launch
extern "C" void kernel_launch(void* const* d_in, const int* in_sizes, int n_in,
                              void* d_out, int out_size, void* d_ws, size_t ws_size,
                              hipStream_t stream) {
  (void)in_sizes; (void)n_in; (void)out_size; (void)ws_size;
  const float*         ndata     = (const float*)d_in[0];
  const float*         attn_bias = (const float*)d_in[1];
  const unsigned char* attn_mask = (const unsigned char*)d_in[2];
  const float* Wq = (const float*)d_in[3];
  const float* bq = (const float*)d_in[4];
  const float* Wk = (const float*)d_in[5];
  const float* bk = (const float*)d_in[6];
  const float* Wv = (const float*)d_in[7];
  const float* bv = (const float*)d_in[8];
  const float* Wo = (const float*)d_in[9];
  const float* bo = (const float*)d_in[10];
  float* out = (float*)d_out;

  const size_t NDE = (size_t)B_ * N_ * FEAT_;   // 2M elements
  const size_t WE  = (size_t)FEAT_ * FEAT_;     // 256K elements
  bf16* nd_bf = (bf16*)d_ws;
  bf16* wq_bf = nd_bf + NDE;
  bf16* wk_bf = wq_bf + WE;
  bf16* wv_bf = wk_bf + WE;
  bf16* wo_bf = wv_bf + WE;
  bf16* Qb    = wo_bf + WE;
  bf16* Kb    = Qb + NDE;
  bf16* Vtb   = Kb + NDE;
  bf16* Ob    = Vtb + NDE;   // total ~22 MB

  cvt_kernel<<<dim3((unsigned)((NDE + 255) / 256)), 256, 0, stream>>>(ndata, nd_bf, (int)NDE);
  cvt_kernel<<<dim3((unsigned)((WE + 255) / 256)), 256, 0, stream>>>(Wq, wq_bf, (int)WE);
  cvt_kernel<<<dim3((unsigned)((WE + 255) / 256)), 256, 0, stream>>>(Wk, wk_bf, (int)WE);
  cvt_kernel<<<dim3((unsigned)((WE + 255) / 256)), 256, 0, stream>>>(Wv, wv_bf, (int)WE);
  cvt_kernel<<<dim3((unsigned)((WE + 255) / 256)), 256, 0, stream>>>(Wo, wo_bf, (int)WE);

  dim3 gq((B_ * N_) / 64, FEAT_ / 16, 3);     // 4 M-tiles per workgroup (4 waves)
  qkv_proj_kernel<<<gq, 128, 0, stream>>>(nd_bf, wq_bf, wk_bf, wv_bf, bq, bk, bv, Qb, Kb, Vtb);

  attn_kernel<<<B_ * H_ * (N_ / 16), 256, 0, stream>>>(Qb, Kb, Vtb, attn_bias, attn_mask, Ob);

  dim3 go((B_ * N_) / 64, FEAT_ / 16);
  out_proj_kernel<<<go, 128, 0, stream>>>(Ob, wo_bf, bo, out);
}
